// Performer_Baseline_1640677507223
// MI455X (gfx1250) — compile-verified, matrix operand order checked by reading, not run on hardware
//
#include <hip/hip_runtime.h>
#include <hip/hip_bf16.h>
#include <math.h>
#include <stdint.h>

typedef __attribute__((ext_vector_type(16))) _Float16 v16h;
typedef __attribute__((ext_vector_type(8)))  _Float16 v8h;
typedef __attribute__((ext_vector_type(8)))  float    v8f;
typedef __attribute__((ext_vector_type(4)))  int      i32x4;
typedef __attribute__((address_space(1)))    i32x4    as1_i32x4;
typedef __attribute__((address_space(3)))    i32x4    as3_i32x4;

namespace {

constexpr int Bb = 128, Cc = 9, Nn = 2048, Dd = 192, Hh = 384, NCc = 18;

// ---------------------------------------------------------------------------
// WMMA helpers (CDNA5 wave32, 16x16x32 f16 -> f32)
// ---------------------------------------------------------------------------
__device__ __forceinline__ v8f wmma_f16(v16h a, v16h b, v8f c) {
  return __builtin_amdgcn_wmma_f32_16x16x32_f16(false, a, false, b, (short)0, c,
                                                false, false);
}

// Load one 16x32 fragment from a row-major tile (ld = row stride in halves).
// A layout (16-bit, 16x32): lane l (l<16) holds row M=l with K in {0..7,16..23};
// lane l+16 holds row M=l with K in {8..15,24..31}. Serves B-fragments too when
// the operand is stored transposed (column n of B == row n of B^T).
__device__ __forceinline__ v16h load_frag(const _Float16* p, int ld) {
  const int lane = threadIdx.x & 31;
  const int r    = lane & 15;
  const int ko   = (lane >> 4) << 3;
  const _Float16* q = p + (size_t)r * ld + ko;
  v8h lo = *(const v8h*)(q);
  v8h hi = *(const v8h*)(q + 16);
  v16h f;
#pragma unroll
  for (int i = 0; i < 8; ++i) { f[i] = lo[i]; f[i + 8] = hi[i]; }
  return f;
}

// ---------------------------------------------------------------------------
// CDNA5 async global->LDS copy (ASYNCcnt path) + wait
// ---------------------------------------------------------------------------
__device__ __forceinline__ void async_copy_b128(const void* g, void* l) {
#if __has_builtin(__builtin_amdgcn_global_load_async_to_lds_b128)
  __builtin_amdgcn_global_load_async_to_lds_b128(
      (as1_i32x4*)(uintptr_t)g,
      (as3_i32x4*)(uint32_t)(uintptr_t)l,   // generic->LDS: low 32 bits
      0, 0);
#else
  uint32_t lofs = (uint32_t)(uintptr_t)l;
  asm volatile("global_load_async_to_lds_b128 %0, %1, off"
               :: "v"(lofs), "v"((uint64_t)(uintptr_t)g) : "memory");
#endif
}

template <int N>
__device__ __forceinline__ void wait_async() {
#if __has_builtin(__builtin_amdgcn_s_wait_asynccnt)
  __builtin_amdgcn_s_wait_asynccnt(N);
#else
  asm volatile("s_wait_asynccnt %0" :: "n"(N) : "memory");
#endif
}

// ---------------------------------------------------------------------------
// Small utility kernels
// ---------------------------------------------------------------------------
__global__ void k_zero(float* p, int n) {
  int i = blockIdx.x * blockDim.x + threadIdx.x;
  if (i < n) p[i] = 0.f;
}

__global__ void k_cvt(const float* __restrict__ s, _Float16* __restrict__ d, int n) {
  int i = blockIdx.x * blockDim.x + threadIdx.x;
  if (i < n) d[i] = (_Float16)s[i];
}

// ---------------------------------------------------------------------------
// Conv1d(C=9 -> D=192, k=3, pad=1) + pos_embed -> h[b,n,d] fp32
// ---------------------------------------------------------------------------
__global__ void k_conv(const float* __restrict__ x, const float* __restrict__ w,
                       const float* __restrict__ cb, const float* __restrict__ pos,
                       float* __restrict__ h) {
  const int n = blockIdx.x, b = blockIdx.y, d = threadIdx.x;
  __shared__ float xs[Cc * 3];
  if (d < Cc * 3) {
    int c = d / 3, t = d % 3, nn = n + t - 1;
    xs[d] = (nn >= 0 && nn < Nn) ? x[((size_t)b * Cc + c) * Nn + nn] : 0.f;
  }
  __syncthreads();
  float acc = cb[d];
#pragma unroll
  for (int c = 0; c < Cc; ++c)
#pragma unroll
    for (int t = 0; t < 3; ++t) acc += w[(d * Cc + c) * 3 + t] * xs[c * 3 + t];
  acc += pos[(size_t)n * Dd + d];
  h[((size_t)b * Nn + n) * Dd + d] = acc;
}

// ---------------------------------------------------------------------------
// LayerNorm over D=192 per token; writes f16. Block (192 thr) per (n,b).
// ---------------------------------------------------------------------------
__global__ void k_ln(const float* __restrict__ h, const float* __restrict__ g,
                     const float* __restrict__ bt, _Float16* __restrict__ out) {
  const int n = blockIdx.x, b = blockIdx.y, d = threadIdx.x;
  __shared__ float s1[Dd], s2[Dd], mv[2];
  const size_t idx = ((size_t)b * Nn + n) * Dd + d;
  const float v = h[idx];
  s1[d] = v; s2[d] = v * v;
  __syncthreads();
  for (int s = Dd / 2; s > 1; s >>= 1) {  // 96,48,24,12,6,3
    if (d < s) { s1[d] += s1[d + s]; s2[d] += s2[d + s]; }
    __syncthreads();
  }
  if (d == 0) {
    float sum = s1[0] + s1[1] + s1[2];
    float sq  = s2[0] + s2[1] + s2[2];
    float m = sum / Dd;
    float var = sq / Dd - m * m;
    mv[0] = m; mv[1] = rsqrtf(var + 1e-5f);
  }
  __syncthreads();
  out[idx] = (_Float16)((v - mv[0]) * mv[1] * g[d] + bt[d]);
}

// ---------------------------------------------------------------------------
// qkv = hn @ qkv_w^T + b ; relu(q), relu(k)
// q f16 row-major [B,N,D]; kT,vT f16 feature-major [B,D,N]; denom[b] atomics.
// Block 128 (4 waves); wave = one 16x16 tile. grid (9, N/16, B).
// ---------------------------------------------------------------------------
__global__ void k_qkv(const _Float16* __restrict__ hn, const _Float16* __restrict__ wq,
                      const float* __restrict__ qb, _Float16* __restrict__ q16,
                      _Float16* __restrict__ kT, _Float16* __restrict__ vT,
                      float* __restrict__ denom) {
  const int wv = threadIdx.x >> 5;
  const int f0 = (blockIdx.x * 4 + wv) * 16;  // 0..560
  const int n0 = blockIdx.y * 16;
  const int b  = blockIdx.z;
  const int lane = threadIdx.x & 31;
  const int nn = lane & 15;
  const int mo = (lane >> 4) << 3;

  v8f acc = {};
  const _Float16* abase = hn + ((size_t)b * Nn + n0) * Dd;
  const _Float16* bbase = wq + (size_t)f0 * Dd;
#pragma unroll
  for (int kc = 0; kc < Dd / 32; ++kc) {
    v16h a  = load_frag(abase + kc * 32, Dd);
    v16h bf = load_frag(bbase + kc * 32, Dd);
    acc = wmma_f16(a, bf, acc);
  }

  const int f = f0 + nn;
  const float bias = qb[f];
  float qsum = 0.f;
  if (f0 < Dd) {                    // Q: relu, row-major store, denom partial
#pragma unroll
    for (int r = 0; r < 8; ++r) {
      float val = fmaxf(acc[r] + bias, 0.f);
      qsum += val;
      q16[((size_t)b * Nn + n0 + mo + r) * Dd + f] = (_Float16)val;
    }
#pragma unroll
    for (int off = 16; off > 0; off >>= 1) qsum += __shfl_xor(qsum, off);
    if (lane == 0) atomicAdd(&denom[b], qsum);
  } else if (f0 < 2 * Dd) {         // K: relu, transposed packed store
    v8h pk;
#pragma unroll
    for (int r = 0; r < 8; ++r) pk[r] = (_Float16)fmaxf(acc[r] + bias, 0.f);
    *(v8h*)(kT + ((size_t)b * Dd + (f - Dd)) * Nn + n0 + mo) = pk;
  } else {                          // V: transposed packed store
    v8h pk;
#pragma unroll
    for (int r = 0; r < 8; ++r) pk[r] = (_Float16)(acc[r] + bias);
    *(v8h*)(vT + ((size_t)b * Dd + (f - 2 * Dd)) * Nn + n0 + mo) = pk;
  }
}

// ---------------------------------------------------------------------------
// kv[d,e] = sum_n k[n,d] v[n,e] (K=2048), stored transposed kvcm[e,d] f16.
// Block 128 (4 waves); wave = one 16x16 tile. grid (36, B).
// ---------------------------------------------------------------------------
__global__ void k_kv(const _Float16* __restrict__ kT, const _Float16* __restrict__ vT,
                     _Float16* __restrict__ kvcm) {
  const int wv = threadIdx.x >> 5;
  const int t  = blockIdx.x * 4 + wv;    // 0..143
  const int d0 = (t / 12) * 16, e0 = (t % 12) * 16;
  const int b  = blockIdx.y;

  v8f acc = {};
  const _Float16* abase = kT + ((size_t)b * Dd + d0) * Nn;
  const _Float16* bbase = vT + ((size_t)b * Dd + e0) * Nn;
  for (int kc = 0; kc < Nn / 32; ++kc) {
    if (kc + 1 < Nn / 32) {
      __builtin_prefetch(abase + (kc + 1) * 32, 0, 1);
      __builtin_prefetch(bbase + (kc + 1) * 32, 0, 1);
    }
    v16h a  = load_frag(abase + kc * 32, Nn);
    v16h bf = load_frag(bbase + kc * 32, Nn);
    acc = wmma_f16(a, bf, acc);
  }
  const int lane = threadIdx.x & 31, nn = lane & 15, mo = (lane >> 4) << 3;
  v8h pk;
#pragma unroll
  for (int r = 0; r < 8; ++r) pk[r] = (_Float16)acc[r];
  *(v8h*)(kvcm + ((size_t)b * Dd + e0 + nn) * Dd + d0 + mo) = pk;  // kv^T
}

// ---------------------------------------------------------------------------
// Fused: num = q @ kv ; attn = num/denom ; h += attn @ proj_w^T + proj_b
// Block = 128 thr (4 waves, 4 token tiles). kv k-slices (192x32 f16) are
// double-buffered in LDS via async global->LDS (ASYNCcnt path), shared by all
// 4 waves. Each wave issues 6 b128 async copies per slice; waiting
// ASYNCcnt<=6 after the next slice is issued guarantees the current slice
// has landed (in-order completion), then a block barrier publishes it.
// grid (N/64, B), block 128.
// ---------------------------------------------------------------------------
__global__ void __launch_bounds__(128) k_attn(
    const _Float16* __restrict__ q16, const _Float16* __restrict__ kvcm,
    const _Float16* __restrict__ wp, const float* __restrict__ pb,
    const float* __restrict__ denom, float* __restrict__ h) {
  const int wv = threadIdx.x >> 5;
  const int n0 = (blockIdx.x * 4 + wv) * 16;
  const int b  = blockIdx.y;
  const int lane = threadIdx.x & 31, nn = lane & 15, mo = (lane >> 4) << 3;

  __shared__ _Float16 kvs[2][Dd][32];   // double-buffered kv k-slice (24.0 KB)
  __shared__ _Float16 at[4][16 * 200];  // per-wave attn strip       (25.6 KB)

  const _Float16* kvb = kvcm + (size_t)b * Dd * Dd;
  // Stage k-slice kc into buffer bi: 192 rows x 64B = 768 b128s, 128 threads.
  auto stage = [&](int kc, int bi) {
#pragma unroll
    for (int i = 0; i < 6; ++i) {
      int t = i * 128 + threadIdx.x;
      int e = t >> 2, c = t & 3;
      async_copy_b128(kvb + (size_t)e * Dd + kc * 32 + c * 8, &kvs[bi][e][c * 8]);
    }
  };

  stage(0, 0);
  stage(1, 1);

  v8f zero = {};
  v8f acc[12];
#pragma unroll
  for (int j = 0; j < 12; ++j) acc[j] = zero;

  const _Float16* abase = q16 + ((size_t)b * Nn + n0) * Dd;
  for (int kc = 0; kc < 6; ++kc) {
    const int bi = kc & 1;
    if (kc < 5) wait_async<6>();   // slice kc retired; kc+1 may be in flight
    else        wait_async<0>();
    __syncthreads();               // publish slice kc to all waves
    v16h a = load_frag(abase + kc * 32, Dd);
#pragma unroll
    for (int j = 0; j < 12; ++j) {
      v16h bf = load_frag(&kvs[bi][j * 16][0], 32);
      acc[j] = wmma_f16(a, bf, acc[j]);
    }
    __syncthreads();               // all waves done reading buffer bi
    if (kc + 2 < 6) stage(kc + 2, bi);
  }

  const float sc = 1.f / (denom[b] + 1e-6f);
#pragma unroll
  for (int j = 0; j < 12; ++j)
#pragma unroll
    for (int r = 0; r < 8; ++r)
      at[wv][(mo + r) * 200 + j * 16 + nn] = (_Float16)(acc[j][r] * sc);
  __syncthreads();

  v8f pacc[12];
#pragma unroll
  for (int j = 0; j < 12; ++j) pacc[j] = zero;
#pragma unroll
  for (int kc = 0; kc < 6; ++kc) {
    v16h a = load_frag(&at[wv][kc * 32], 200);
#pragma unroll
    for (int j = 0; j < 12; ++j) {
      v16h bf = load_frag(wp + (size_t)(j * 16) * Dd + kc * 32, Dd);
      pacc[j] = wmma_f16(a, bf, pacc[j]);
    }
  }
#pragma unroll
  for (int j = 0; j < 12; ++j) {
    const int f = j * 16 + nn;
    const float bias = pb[f];
#pragma unroll
    for (int r = 0; r < 8; ++r) {
      const size_t idx = ((size_t)b * Nn + n0 + mo + r) * Dd + f;
      h[idx] += pacc[j][r] + bias;  // in-place residual (disjoint lanes)
    }
  }
}

// ---------------------------------------------------------------------------
// m = gelu(hn2 @ mlp_w1^T + b1), exact GELU. grid (6, N/16, B), block 128.
// ---------------------------------------------------------------------------
__global__ void k_mlp1(const _Float16* __restrict__ hn2, const _Float16* __restrict__ w1,
                       const float* __restrict__ b1, _Float16* __restrict__ m16) {
  const int wv = threadIdx.x >> 5;
  const int f0 = (blockIdx.x * 4 + wv) * 16;  // 0..368
  const int n0 = blockIdx.y * 16, b = blockIdx.z;
  v8f acc = {};
  const _Float16* abase = hn2 + ((size_t)b * Nn + n0) * Dd;
  const _Float16* bbase = w1 + (size_t)f0 * Dd;
#pragma unroll
  for (int kc = 0; kc < Dd / 32; ++kc) {
    v16h a  = load_frag(abase + kc * 32, Dd);
    v16h bf = load_frag(bbase + kc * 32, Dd);
    acc = wmma_f16(a, bf, acc);
  }
  const int lane = threadIdx.x & 31, nn = lane & 15, mo = (lane >> 4) << 3;
  const int f = f0 + nn;
  const float bias = b1[f];
#pragma unroll
  for (int r = 0; r < 8; ++r) {
    float val = acc[r] + bias;
    float ge  = 0.5f * val * (1.f + erff(val * 0.70710678118f));
    m16[((size_t)b * Nn + n0 + mo + r) * Hh + f] = (_Float16)ge;
  }
}

// ---------------------------------------------------------------------------
// h3 = h + m @ mlp_w2^T + b2, reduced straight into pooled[b,f].
// grid (3, N/16, B), block 128.
// ---------------------------------------------------------------------------
__global__ void k_mlp2(const _Float16* __restrict__ m16, const _Float16* __restrict__ w2,
                       const float* __restrict__ b2, const float* __restrict__ h,
                       float* __restrict__ pooled) {
  const int wv = threadIdx.x >> 5;
  const int f0 = (blockIdx.x * 4 + wv) * 16;  // 0..176
  const int n0 = blockIdx.y * 16, b = blockIdx.z;
  v8f acc = {};
  const _Float16* abase = m16 + ((size_t)b * Nn + n0) * Hh;
  const _Float16* bbase = w2 + (size_t)f0 * Hh;
#pragma unroll
  for (int kc = 0; kc < Hh / 32; ++kc) {
    v16h a  = load_frag(abase + kc * 32, Hh);
    v16h bf = load_frag(bbase + kc * 32, Hh);
    acc = wmma_f16(a, bf, acc);
  }
  const int lane = threadIdx.x & 31, nn = lane & 15, mo = (lane >> 4) << 3;
  const int f = f0 + nn;
  const float bias = b2[f];
  float ps = 0.f;
#pragma unroll
  for (int r = 0; r < 8; ++r) {
    const size_t idx = ((size_t)b * Nn + n0 + mo + r) * Dd + f;
    ps += h[idx] + acc[r] + bias;
  }
  atomicAdd(&pooled[(size_t)b * Dd + f], ps);
}

// ---------------------------------------------------------------------------
// out[b,c] = (pooled[b,:]/N) . head_w[c,:] + head_b[c]
// ---------------------------------------------------------------------------
__global__ void k_head(const float* __restrict__ pooled, const float* __restrict__ hw,
                       const float* __restrict__ hb, float* __restrict__ out) {
  const int b = blockIdx.x, c = threadIdx.x;
  if (c < NCc) {
    float s = hb[c];
    const float invN = 1.f / Nn;
    for (int d = 0; d < Dd; ++d)
      s += pooled[(size_t)b * Dd + d] * invN * hw[c * Dd + d];
    out[b * NCc + c] = s;
  }
}

}  // namespace

extern "C" void kernel_launch(void* const* d_in, const int* in_sizes, int n_in,
                              void* d_out, int out_size, void* d_ws, size_t ws_size,
                              hipStream_t stream) {
  (void)in_sizes; (void)n_in; (void)out_size; (void)ws_size;

  const float* x      = (const float*)d_in[0];
  const float* conv_w = (const float*)d_in[1];
  const float* conv_b = (const float*)d_in[2];
  const float* pos    = (const float*)d_in[3];
  const float* qkv_w  = (const float*)d_in[4];
  const float* qkv_b  = (const float*)d_in[5];
  const float* proj_w = (const float*)d_in[6];
  const float* proj_b = (const float*)d_in[7];
  const float* ln1_g  = (const float*)d_in[8];
  const float* ln1_b  = (const float*)d_in[9];
  const float* mlp_w1 = (const float*)d_in[10];
  const float* mlp_b1 = (const float*)d_in[11];
  const float* mlp_w2 = (const float*)d_in[12];
  const float* mlp_b2 = (const float*)d_in[13];
  const float* ln2_g  = (const float*)d_in[14];
  const float* ln2_b  = (const float*)d_in[15];
  const float* head_w = (const float*)d_in[16];
  const float* head_b = (const float*)d_in[17];
  float* out = (float*)d_out;

  char* wp_ = (char*)d_ws;
  auto carve = [&](size_t bytes) -> char* {
    char* p = wp_;
    wp_ += (bytes + 255) & ~(size_t)255;
    return p;
  };
  float*    h      = (float*)   carve((size_t)Bb * Nn * Dd * 4);  // residual, in-place
  _Float16* hn16   = (_Float16*)carve((size_t)Bb * Nn * Dd * 2);  // hn, then hn2
  _Float16* q16    = (_Float16*)carve((size_t)Bb * Nn * Dd * 2);
  _Float16* kT16   = (_Float16*)carve((size_t)Bb * Nn * Dd * 2);
  _Float16* vT16   = (_Float16*)carve((size_t)Bb * Nn * Dd * 2);
  _Float16* m16    = kT16;  // reuse kT+vT region (2 * 96 MiB == B*N*H*2)
  _Float16* kvcm   = (_Float16*)carve((size_t)Bb * Dd * Dd * 2);
  float*    denom  = (float*)   carve((size_t)Bb * 4);
  float*    pooled = (float*)   carve((size_t)Bb * Dd * 4);
  _Float16* wq16   = (_Float16*)carve((size_t)3 * Dd * Dd * 2);
  _Float16* wp16   = (_Float16*)carve((size_t)Dd * Dd * 2);
  _Float16* w116   = (_Float16*)carve((size_t)Hh * Dd * 2);
  _Float16* w216   = (_Float16*)carve((size_t)Dd * Hh * 2);

  // weight conversion + accumulator init
  k_cvt<<<(3 * Dd * Dd + 255) / 256, 256, 0, stream>>>(qkv_w, wq16, 3 * Dd * Dd);
  k_cvt<<<(Dd * Dd + 255) / 256, 256, 0, stream>>>(proj_w, wp16, Dd * Dd);
  k_cvt<<<(Hh * Dd + 255) / 256, 256, 0, stream>>>(mlp_w1, w116, Hh * Dd);
  k_cvt<<<(Dd * Hh + 255) / 256, 256, 0, stream>>>(mlp_w2, w216, Dd * Hh);
  k_zero<<<1, 256, 0, stream>>>(denom, Bb);
  k_zero<<<(Bb * Dd + 255) / 256, 256, 0, stream>>>(pooled, Bb * Dd);

  // patch embed + LN1
  k_conv<<<dim3(Nn, Bb), Dd, 0, stream>>>(x, conv_w, conv_b, pos, h);
  k_ln<<<dim3(Nn, Bb), Dd, 0, stream>>>(h, ln1_g, ln1_b, hn16);

  // attention (linear / Performer-style)
  k_qkv<<<dim3(9, Nn / 16, Bb), 128, 0, stream>>>(hn16, wq16, qkv_b, q16, kT16, vT16, denom);
  k_kv<<<dim3(36, Bb), 128, 0, stream>>>(kT16, vT16, kvcm);
  k_attn<<<dim3(Nn / 64, Bb), 128, 0, stream>>>(q16, kvcm, wp16, proj_b, denom, h);

  // MLP block + pooled reduction
  k_ln<<<dim3(Nn, Bb), Dd, 0, stream>>>(h, ln2_g, ln2_b, hn16);
  k_mlp1<<<dim3(6, Nn / 16, Bb), 128, 0, stream>>>(hn16, w116, mlp_b1, m16);
  k_mlp2<<<dim3(3, Nn / 16, Bb), 128, 0, stream>>>(m16, w216, mlp_b2, h, pooled);

  // classifier head
  k_head<<<dim3(Bb), 32, 0, stream>>>(pooled, head_w, head_b, out);
}
